// GNNOnly_37357625541087
// MI455X (gfx1250) — matrix operand order repaired and checked by reading, not compiled, not run on hardware
//
#include <hip/hip_runtime.h>
#include <hip/hip_bf16.h>

// Problem constants (match reference)
#define B_   8
#define T_   8
#define N_   5000
#define FIN  128
#define DM   256
#define NN_  40000      // B_*N_
#define E_   640000

typedef __attribute__((ext_vector_type(16))) __bf16 v16bf;
typedef __attribute__((ext_vector_type(8)))  __bf16 v8bf;
typedef __attribute__((ext_vector_type(8)))  float  v8f;

// ---------------------------------------------------------------------------
// x = snapshots[:, T-1].reshape(NN, FIN)  converted to bf16 (row-major)
// ---------------------------------------------------------------------------
__global__ void k_convert_x(const float* __restrict__ snaps, __bf16* __restrict__ xbf) {
    int g = blockIdx.x * blockDim.x + threadIdx.x;
    if (g >= NN_ * FIN) return;
    int i = g >> 7;            // node id in [0, NN_)
    int f = g & (FIN - 1);
    int b = i / N_;
    int n = i - b * N_;
    size_t off = ((size_t)(b * T_ + (T_ - 1)) * N_ + n) * FIN + f;
    xbf[g] = (__bf16)snaps[off];
}

// W [K, Dout] f32 row-major  ->  Wt [Dout, K] bf16 row-major (B-fragment friendly)
__global__ void k_convert_wT(const float* __restrict__ W, __bf16* __restrict__ Wt,
                             int K, int Dout) {
    int g = blockIdx.x * blockDim.x + threadIdx.x;
    if (g >= K * Dout) return;
    int d = g / K;
    int k = g - d * K;
    Wt[g] = (__bf16)W[(size_t)k * Dout + d];
}

__global__ void k_fill_f32(float* __restrict__ p, float v, int n) {
    int g = blockIdx.x * blockDim.x + threadIdx.x;
    if (g < n) p[g] = v;
}

__global__ void k_zero_f4(float4* __restrict__ p, int n4) {
    int g = blockIdx.x * blockDim.x + threadIdx.x;
    if (g < n4) p[g] = make_float4(0.f, 0.f, 0.f, 0.f);
}

// deg accumulation over real edges (self-loop pre-seeded as 1.0)
__global__ void k_deg(const int* __restrict__ dst, float* __restrict__ deg, int e) {
    int g = blockIdx.x * blockDim.x + threadIdx.x;
    if (g < e) atomicAdd(&deg[dst[g]], 1.0f);
}

__global__ void k_rsqrt_inplace(float* __restrict__ p, int n) {
    int g = blockIdx.x * blockDim.x + threadIdx.x;
    if (g < n) {
        float d = p[g];
        p[g] = (d > 0.f) ? rsqrtf(d) : 0.f;
    }
}

// ---------------------------------------------------------------------------
// WMMA GEMM:  C[M, DM] = A[M, K](bf16) @ Bt[DM, K]^T(bf16), f32 accumulate.
// Block = 256 threads = 8 waves; wave (wm, wn) owns a 16x32 strip; block tile
// is 64x64.  Grid = (M/64, DM/64).  K in {128, 256}, unrolled.
// A/B use contiguous k order (k = 16h + s, same in both); C/D per CDNA5 ISA 7.12.2 (wave32).
// ---------------------------------------------------------------------------
template <int K>
__global__ __launch_bounds__(256) void k_wmma_gemm(const __bf16* __restrict__ A,
                                                   const __bf16* __restrict__ Bt,
                                                   float* __restrict__ C) {
    const int lane   = threadIdx.x & 31;
    const int w      = threadIdx.x >> 5;
    const int wm     = w & 3;           // 4 M sub-tiles
    const int wn     = w >> 2;          // 2 N sub-tiles (32 cols each)
    const int hiHalf = lane >> 4;       // 0: lanes 0-15, 1: lanes 16-31
    const int rowA   = blockIdx.x * 64 + wm * 16 + (lane & 15);
    const int n0     = blockIdx.y * 64 + wn * 32;
    const int colB0  = n0 + (lane & 15);

    const __bf16* aRow  = A  + (size_t)rowA * K;
    const __bf16* b0Row = Bt + (size_t)colB0 * K;
    const __bf16* b1Row = b0Row + (size_t)16 * K;

    v8f acc0 = {};
    v8f acc1 = {};

#pragma unroll
    for (int kb = 0; kb < K; kb += 32) {
        // A fragment: lane<16 holds K kb..kb+15 of its row; lane>=16: +16
        v16bf a  = *(const v16bf*)(aRow + kb + hiHalf * 16);
        // B fragment: lane<16 holds K kb..kb+15 of its column; lane>=16: +16
        v16bf b0 = *(const v16bf*)(b0Row + kb + hiHalf * 16);
        v16bf b1 = *(const v16bf*)(b1Row + kb + hiHalf * 16);

        acc0 = __builtin_amdgcn_wmma_f32_16x16x32_bf16(false, a, false, b0,
                                                       (short)0, acc0, false, false);
        acc1 = __builtin_amdgcn_wmma_f32_16x16x32_bf16(false, a, false, b1,
                                                       (short)0, acc1, false, false);
    }

    // C/D layout: VGPR v -> row v (+8 for lanes 16-31), col = lane&15
    const int rBase = blockIdx.x * 64 + wm * 16 + hiHalf * 8;
    const int c0    = n0 + (lane & 15);
#pragma unroll
    for (int v = 0; v < 8; ++v) {
        C[(size_t)(rBase + v) * DM + c0]      = acc0[v];
        C[(size_t)(rBase + v) * DM + c0 + 16] = acc1[v];
    }
}

// ---------------------------------------------------------------------------
// Edge scatter: agg[dst] += dinv[src]*dinv[dst] * h[src]   (one block per edge)
// ---------------------------------------------------------------------------
__global__ __launch_bounds__(DM) void k_scatter(const int* __restrict__ src,
                                                const int* __restrict__ dst,
                                                const float* __restrict__ dinv,
                                                const float* __restrict__ h,
                                                float* __restrict__ agg) {
    int e = blockIdx.x;
    int s = src[e];
    int d = dst[e];
    float coef = dinv[s] * dinv[d];
    int f = threadIdx.x;
    atomicAdd(&agg[(size_t)d * DM + f], coef * h[(size_t)s * DM + f]);
}

// Layer-1 epilogue: add self-loop term + bias, ReLU, convert to bf16 for GEMM2
__global__ __launch_bounds__(DM) void k_epilogue1(const float* __restrict__ agg,
                                                  const float* __restrict__ h,
                                                  const float* __restrict__ dinv,
                                                  const float* __restrict__ bias,
                                                  __bf16* __restrict__ x2) {
    int i = blockIdx.x;
    int f = threadIdx.x;
    float di = dinv[i];
    float v = agg[(size_t)i * DM + f] + h[(size_t)i * DM + f] * di * di + bias[f];
    x2[(size_t)i * DM + f] = (__bf16)fmaxf(v, 0.f);
}

// Mean pool with fused layer-2 epilogue (self-loop + bias + ReLU)
__global__ __launch_bounds__(DM) void k_pool(const float* __restrict__ agg,
                                             const float* __restrict__ h,
                                             const float* __restrict__ dinv,
                                             const float* __restrict__ bias,
                                             float* __restrict__ pooled) {
    int b = blockIdx.x;
    int f = threadIdx.x;
    float bi = bias[f];
    float s = 0.f;
    for (int n = 0; n < N_; ++n) {
        int i = b * N_ + n;
        float di = dinv[i];
        float v = agg[(size_t)i * DM + f] + h[(size_t)i * DM + f] * di * di + bi;
        s += fmaxf(v, 0.f);
    }
    pooled[b * DM + f] = s * (1.0f / (float)N_);
}

// MLP head: relu(pooled@Wh1+bh1) -> softplus(@Wh2+bh2)+1e-6, broadcast over T
__global__ __launch_bounds__(256) void k_head(const float* __restrict__ pooled,
                                              const float* __restrict__ Wh1,
                                              const float* __restrict__ bh1,
                                              const float* __restrict__ Wh2,
                                              const float* __restrict__ bh2,
                                              float* __restrict__ out) {
    __shared__ float hh[B_ * 64];
    int tid = threadIdx.x;
    for (int idx = tid; idx < B_ * 64; idx += blockDim.x) {
        int b = idx >> 6;
        int j = idx & 63;
        float s = bh1[j];
        for (int k = 0; k < DM; ++k) s += pooled[b * DM + k] * Wh1[k * 64 + j];
        hh[idx] = fmaxf(s, 0.f);
    }
    __syncthreads();
    if (tid < B_ * 2) {
        int b = tid >> 1;
        int c = tid & 1;
        float s = bh2[c];
        for (int j = 0; j < 64; ++j) s += hh[b * 64 + j] * Wh2[j * 2 + c];
        float sp = (s > 20.f) ? s : log1pf(expf(s));
        sp += 1e-6f;
        for (int t = 0; t < T_; ++t) out[((size_t)b * T_ + t) * 2 + c] = sp;
    }
}

// ---------------------------------------------------------------------------
extern "C" void kernel_launch(void* const* d_in, const int* in_sizes, int n_in,
                              void* d_out, int out_size, void* d_ws, size_t ws_size,
                              hipStream_t stream) {
    (void)in_sizes; (void)n_in; (void)out_size; (void)ws_size;

    const float* snaps = (const float*)d_in[0];
    const int*   eidx  = (const int*)d_in[1];
    const float* W1    = (const float*)d_in[2];
    const float* b1    = (const float*)d_in[3];
    const float* W2    = (const float*)d_in[4];
    const float* b2    = (const float*)d_in[5];
    const float* Wh1   = (const float*)d_in[6];
    const float* bh1   = (const float*)d_in[7];
    const float* Wh2   = (const float*)d_in[8];
    const float* bh2   = (const float*)d_in[9];
    float* out = (float*)d_out;

    const int* src = eidx;        // edge_index[0]
    const int* dst = eidx + E_;   // edge_index[1]

    // Workspace carve-out (256-byte aligned slabs, ~113 MB total)
    char* ws = (char*)d_ws;
    size_t off = 0;
    auto carve = [&](size_t bytes) -> void* {
        void* p = ws + off;
        off = (off + bytes + 255) & ~(size_t)255;
        return p;
    };
    __bf16* xbf    = (__bf16*)carve((size_t)NN_ * FIN * sizeof(__bf16));
    __bf16* w1t    = (__bf16*)carve((size_t)DM * FIN * sizeof(__bf16));
    __bf16* w2t    = (__bf16*)carve((size_t)DM * DM * sizeof(__bf16));
    float*  h      = (float*) carve((size_t)NN_ * DM * sizeof(float));
    float*  agg    = (float*) carve((size_t)NN_ * DM * sizeof(float));
    float*  dinv   = (float*) carve((size_t)NN_ * sizeof(float));
    __bf16* x2bf   = (__bf16*)carve((size_t)NN_ * DM * sizeof(__bf16));
    float*  pooled = (float*) carve((size_t)B_ * DM * sizeof(float));

    const int n4 = NN_ * DM / 4;

    // ---- prep: bf16 inputs, transposed weights, degree normalization ----
    k_convert_x<<<(NN_ * FIN + 255) / 256, 256, 0, stream>>>(snaps, xbf);
    k_convert_wT<<<(FIN * DM + 255) / 256, 256, 0, stream>>>(W1, w1t, FIN, DM);
    k_convert_wT<<<(DM * DM + 255) / 256, 256, 0, stream>>>(W2, w2t, DM, DM);
    k_fill_f32<<<(NN_ + 255) / 256, 256, 0, stream>>>(dinv, 1.0f, NN_); // self-loop
    k_deg<<<(E_ + 255) / 256, 256, 0, stream>>>(dst, dinv, E_);
    k_rsqrt_inplace<<<(NN_ + 255) / 256, 256, 0, stream>>>(dinv, NN_);

    // ---- layer 1: WMMA GEMM + scatter + epilogue ----
    k_wmma_gemm<FIN><<<dim3(NN_ / 64, DM / 64), 256, 0, stream>>>(xbf, w1t, h);
    k_zero_f4<<<(n4 + 255) / 256, 256, 0, stream>>>((float4*)agg, n4);
    k_scatter<<<E_, DM, 0, stream>>>(src, dst, dinv, h, agg);
    k_epilogue1<<<NN_, DM, 0, stream>>>(agg, h, dinv, b1, x2bf);

    // ---- layer 2: WMMA GEMM + scatter ----
    k_wmma_gemm<DM><<<dim3(NN_ / 64, DM / 64), 256, 0, stream>>>(x2bf, w2t, h);
    k_zero_f4<<<(n4 + 255) / 256, 256, 0, stream>>>((float4*)agg, n4);
    k_scatter<<<E_, DM, 0, stream>>>(src, dst, dinv, h, agg);

    // ---- pool + head ----
    k_pool<<<B_, DM, 0, stream>>>(agg, h, dinv, b2, pooled);
    k_head<<<1, 256, 0, stream>>>(pooled, Wh1, bh1, Wh2, bh2, out);
}
